// TransformerBlock_2516850835983
// MI455X (gfx1250) — matrix operand order repaired and checked by reading, not compiled, not run on hardware
//
#include <hip/hip_runtime.h>
#include <hip/hip_bf16.h>
#include <math.h>

typedef __bf16 bf16;
typedef __attribute__((ext_vector_type(16))) __bf16 v16bf;
typedef __attribute__((ext_vector_type(8)))  float  v8f;
typedef __attribute__((ext_vector_type(4)))  int    v4i;

#define AS1 __attribute__((address_space(1)))
#define AS3 __attribute__((address_space(3)))

#if defined(__HIP_DEVICE_COMPILE__) && \
    __has_builtin(__builtin_amdgcn_global_load_async_to_lds_b128) && \
    __has_builtin(__builtin_amdgcn_s_wait_asynccnt)
#define HAVE_ASYNC_LDS 1
// dst/src pointee type is int4 per the builtin prototype (AS1 global, AS3 LDS)
#define ASYNC_CP_B128(gptr, lptr)                                   \
    __builtin_amdgcn_global_load_async_to_lds_b128(                 \
        (AS1 v4i*)(gptr), (AS3 v4i*)(lptr), 0, 0)
#else
#define HAVE_ASYNC_LDS 0
#endif

union AFrag { unsigned int u[8]; v16bf v; };

__device__ __forceinline__ v8f vzero8() {
    v8f z = {0.f,0.f,0.f,0.f,0.f,0.f,0.f,0.f};
    return z;
}

// ---------------------------------------------------------------------------
// fp32 -> bf16 conversion (weights)
// ---------------------------------------------------------------------------
__global__ __launch_bounds__(256) void cvt_f32_bf16_kernel(
    const float* __restrict__ in, bf16* __restrict__ out, int n) {
    int i = blockIdx.x * 256 + threadIdx.x;
    if (i < n) out[i] = (bf16)in[i];
}

// ---------------------------------------------------------------------------
// LayerNorm -> bf16.  One 256-thread block per row of C=1024.
// ---------------------------------------------------------------------------
__global__ __launch_bounds__(256) void ln_bf16_kernel(
    const float* __restrict__ x, const float* __restrict__ g,
    const float* __restrict__ bta, bf16* __restrict__ out, int C) {
    const int row  = blockIdx.x;
    const float* xr = x + (size_t)row * C;
    float s = 0.f, ss = 0.f;
    for (int i = threadIdx.x; i < C; i += 256) {
        float v = xr[i];
        s += v; ss += v * v;
    }
    #pragma unroll
    for (int off = 16; off > 0; off >>= 1) {
        s  += __shfl_xor(s,  off, 32);
        ss += __shfl_xor(ss, off, 32);
    }
    __shared__ float rs[8], rss[8];
    if ((threadIdx.x & 31) == 0) { rs[threadIdx.x >> 5] = s; rss[threadIdx.x >> 5] = ss; }
    __syncthreads();
    s = 0.f; ss = 0.f;
    #pragma unroll
    for (int w = 0; w < 8; w++) { s += rs[w]; ss += rss[w]; }
    const float mu   = s / (float)C;
    const float var  = ss / (float)C - mu * mu;
    const float rstd = rsqrtf(var + 1e-5f);
    for (int i = threadIdx.x; i < C; i += 256)
        out[(size_t)row * C + i] = (bf16)((xr[i] - mu) * rstd * g[i] + bta[i]);
}

// ---------------------------------------------------------------------------
// Generic bf16 GEMM:  out = act(A[M,K] @ W[K,N] + bias) (+ resid)
// Block tile 128x128, K-step 32, 8 waves (2x4), wave tile 64x32 (4x2 WMMA).
// Double-buffered LDS; A staged with GLOBAL_LOAD_ASYNC_TO_LDS_B128
// (ASYNCcnt overlap of next tile with current compute).
// ---------------------------------------------------------------------------
template<int ACT, bool OUT_BF16, bool RESID>
__global__ __launch_bounds__(256) void gemm_bf16_kernel(
    const bf16* __restrict__ A, const bf16* __restrict__ W,
    const float* __restrict__ bias, const float* __restrict__ resid,
    void* __restrict__ outp, int M, int N, int K) {
    __shared__ __align__(16) bf16 As[2][128][40];   // stride 80B: 16B-aligned rows
    __shared__ __align__(16) bf16 Bs[2][128][34];   // TRANSPOSED: Bs[n][k]

    const int tid  = threadIdx.x;
    const int lane = tid & 31, wid = tid >> 5;
    const int wm   = wid >> 2, wn = wid & 3;          // 2 x 4 wave grid
    const int m0   = blockIdx.y * 128, n0 = blockIdx.x * 128;
    const int hl   = lane >> 4, l16 = lane & 15;

    v8f acc[4][2];
    #pragma unroll
    for (int i = 0; i < 4; i++)
        #pragma unroll
        for (int j = 0; j < 2; j++) acc[i][j] = vzero8();

    const int arow = tid >> 1, ahalf = tid & 1;   // A loader: 16 bf16 each
    const int bk   = tid >> 3, bseg  = tid & 7;   // B loader: 16 bf16 each

    auto issueA = [&](int k0, int buf) {
        const bf16* gsrc = A + (size_t)(m0 + arow) * K + k0 + ahalf * 16;
        bf16* ldst = &As[buf][arow][ahalf * 16];
#if HAVE_ASYNC_LDS
        ASYNC_CP_B128(gsrc, ldst);
        ASYNC_CP_B128(gsrc + 8, ldst + 8);
#else
        const uint4* ag = reinterpret_cast<const uint4*>(gsrc);
        union { uint4 q[2]; unsigned int u[8]; } au;
        au.q[0] = ag[0]; au.q[1] = ag[1];
        unsigned int* dst = reinterpret_cast<unsigned int*>(ldst);
        #pragma unroll
        for (int e = 0; e < 8; e++) dst[e] = au.u[e];
#endif
    };
    auto issueB = [&](int k0, int buf) {
        const uint4* bg = reinterpret_cast<const uint4*>(
            W + (size_t)(k0 + bk) * N + n0 + bseg * 16);
        union { uint4 q[2]; bf16 h[16]; } bu;
        bu.q[0] = bg[0]; bu.q[1] = bg[1];
        #pragma unroll
        for (int e = 0; e < 16; e++) Bs[buf][bseg * 16 + e][bk] = bu.h[e];
    };

    issueA(0, 0);
    issueB(0, 0);
    const int nk = K >> 5;
    for (int it = 0; it < nk; it++) {
        const int cur = it & 1;
#if HAVE_ASYNC_LDS
        __builtin_amdgcn_s_wait_asynccnt(0);
#endif
        __syncthreads();
        if (it + 1 < nk) {
            __builtin_prefetch(W + (size_t)((it + 1) * 32 + bk) * N + n0 + bseg * 16, 0, 1);
            issueA((it + 1) * 32, cur ^ 1);
            issueB((it + 1) * 32, cur ^ 1);
        }

        // ---- B fragments (2 per wave) ----
        v16bf bfrag[2];
        #pragma unroll
        for (int jn = 0; jn < 2; jn++) {
            AFrag f;
            const int col = wn * 32 + jn * 16 + l16;
            const int kb  = hl * 16;
            #pragma unroll
            for (int i = 0; i < 8; i++)
                f.u[i] = *reinterpret_cast<const unsigned int*>(&Bs[cur][col][kb + 2 * i]);
            bfrag[jn] = f.v;
        }
        // ---- A fragments + WMMA ----
        #pragma unroll
        for (int im = 0; im < 4; im++) {
            AFrag f;
            const int row = wm * 64 + im * 16 + l16;
            #pragma unroll
            for (int i = 0; i < 8; i++) {
                const int kk = hl * 16 + 2 * i;
                f.u[i] = *reinterpret_cast<const unsigned int*>(&As[cur][row][kk]);
            }
            #pragma unroll
            for (int jn = 0; jn < 2; jn++)
                acc[im][jn] = __builtin_amdgcn_wmma_f32_16x16x32_bf16(
                    false, f.v, false, bfrag[jn], (short)0, acc[im][jn], false, false);
        }
    }

    // ---- epilogue ----
    #pragma unroll
    for (int im = 0; im < 4; im++) {
        #pragma unroll
        for (int jn = 0; jn < 2; jn++) {
            const int col = n0 + wn * 32 + jn * 16 + l16;
            const float bs = bias[col];
            #pragma unroll
            for (int r = 0; r < 8; r++) {
                const int row = m0 + wm * 64 + im * 16 + hl * 8 + r;
                float v = acc[im][jn][r] + bs;
                if (ACT == 1) v = 0.5f * v * (1.0f + erff(v * 0.70710678118654752f));
                if constexpr (RESID) v += resid[(size_t)row * N + col];
                if constexpr (OUT_BF16)
                    ((bf16*)outp)[(size_t)row * N + col] = (bf16)v;
                else
                    ((float*)outp)[(size_t)row * N + col] = v;
            }
        }
    }
}

// ---------------------------------------------------------------------------
// Flash attention.  qkv: bf16 [B,T,3,H,64] flat = [(b*T+t)*3072 + g*1024 + h*64 + d].
// Grid: (T/128, B*H).  8 waves; wave w owns q rows [tile*128+w*16, +16).
// K tile staged via async-to-LDS; V tile transposed through VGPRs.
// ---------------------------------------------------------------------------
__global__ __launch_bounds__(256) void attn_kernel(
    const bf16* __restrict__ qkv, bf16* __restrict__ attn_out) {
    const int bh   = blockIdx.y;
    const int b    = bh >> 4, hd = bh & 15;
    const int tile = blockIdx.x;
    const int lane = threadIdx.x & 31, wid = threadIdx.x >> 5;
    const int hl   = lane >> 4, l16 = lane & 15;
    const int q0   = tile * 128 + wid * 16;

    __shared__ __align__(16) bf16 Kt[32][72];        // [key][dk]; 144B rows, 16B-aligned
    __shared__ __align__(16) bf16 Vt[64][34];        // TRANSPOSED [dk][key]
    __shared__ __align__(16) bf16 Ps[8][16][34];     // per-wave P relayout scratch

    // ---- load Q fragments (scaled by 1/sqrt(dk)) ----
    v16bf qf[2];
    {
        const int row = q0 + l16;
        const bf16* qrow = qkv + ((size_t)(b * 2048 + row)) * 3072 + hd * 64;
        #pragma unroll
        for (int f = 0; f < 2; f++) {
            v16bf qv;
            #pragma unroll
            for (int i = 0; i < 8; i++) {
                const int kk = hl * 16 + 2 * i + f * 32;
                qv[2 * i]     = (bf16)((float)qrow[kk]     * 0.125f);
                qv[2 * i + 1] = (bf16)((float)qrow[kk + 1] * 0.125f);
            }
            qf[f] = qv;
        }
    }

    v8f o[4];
    #pragma unroll
    for (int j = 0; j < 4; j++) o[j] = vzero8();
    float m_r[8], l_r[8];
    #pragma unroll
    for (int r = 0; r < 8; r++) { m_r[r] = -__builtin_inff(); l_r[r] = 0.f; }

    const int nkb  = tile * 4 + 4;          // key blocks of 32 up to tile end
    const int kkey = threadIdx.x >> 3;      // 0..31 key loader
    const int kseg = threadIdx.x & 7;       // 8-elem d segment

    for (int kb = 0; kb < nkb; kb++) {
        // ---- stage K (async row-major) and V (VGPR transpose) tiles ----
        {
            const size_t krow = ((size_t)(b * 2048 + kb * 32 + kkey)) * 3072;
            const bf16* gk = qkv + krow + 1024 + hd * 64 + kseg * 8;
#if HAVE_ASYNC_LDS
            ASYNC_CP_B128(gk, &Kt[kkey][kseg * 8]);
#else
            *reinterpret_cast<uint4*>(&Kt[kkey][kseg * 8]) =
                *reinterpret_cast<const uint4*>(gk);
#endif
            const uint4 vv = *reinterpret_cast<const uint4*>(qkv + krow + 2048 + hd * 64 + kseg * 8);
            union { uint4 q; bf16 hh[8]; } vu; vu.q = vv;
            #pragma unroll
            for (int e = 0; e < 8; e++) Vt[kseg * 8 + e][kkey] = vu.hh[e];
        }
#if HAVE_ASYNC_LDS
        __builtin_amdgcn_s_wait_asynccnt(0);
#endif
        __syncthreads();

        // ---- S = Q K^T : two 16x16 tiles, 2 WMMAs each over dk ----
        v8f sc[2];
        #pragma unroll
        for (int sub = 0; sub < 2; sub++) {
            v8f s = vzero8();
            const int key = sub * 16 + l16;
            #pragma unroll
            for (int dkb = 0; dkb < 2; dkb++) {
                AFrag f;
                const int kbb = dkb * 32 + hl * 16;
                #pragma unroll
                for (int i = 0; i < 8; i++)
                    f.u[i] = *reinterpret_cast<const unsigned int*>(&Kt[key][kbb + 2 * i]);
                s = __builtin_amdgcn_wmma_f32_16x16x32_bf16(
                        false, qf[dkb], false, f.v, (short)0, s, false, false);
            }
            sc[sub] = s;
        }

        // ---- causal mask + online softmax ----
        float alpha[8];
        #pragma unroll
        for (int r = 0; r < 8; r++) {
            const int qrow = q0 + hl * 8 + r;
            #pragma unroll
            for (int sub = 0; sub < 2; sub++) {
                const int key = kb * 32 + sub * 16 + l16;
                if (key > qrow) sc[sub][r] = -__builtin_inff();
            }
            float mx = fmaxf(sc[0][r], sc[1][r]);
            #pragma unroll
            for (int off = 1; off < 16; off <<= 1) mx = fmaxf(mx, __shfl_xor(mx, off, 32));
            const float mnew = fmaxf(m_r[r], mx);
            const float p0 = expf(sc[0][r] - mnew);
            const float p1 = expf(sc[1][r] - mnew);
            sc[0][r] = p0; sc[1][r] = p1;
            float rsum = p0 + p1;
            #pragma unroll
            for (int off = 1; off < 16; off <<= 1) rsum += __shfl_xor(rsum, off, 32);
            alpha[r] = expf(m_r[r] - mnew);
            l_r[r]   = l_r[r] * alpha[r] + rsum;
            m_r[r]   = mnew;
        }
        #pragma unroll
        for (int j = 0; j < 4; j++)
            #pragma unroll
            for (int r = 0; r < 8; r++) o[j][r] *= alpha[r];

        // ---- relayout P (C-frag -> A-frag) through wave-private LDS ----
        #pragma unroll
        for (int sub = 0; sub < 2; sub++)
            #pragma unroll
            for (int r = 0; r < 8; r++)
                Ps[wid][hl * 8 + r][sub * 16 + l16] = (bf16)sc[sub][r];
        asm volatile("s_wait_dscnt 0" ::: "memory");
        AFrag pf;
        #pragma unroll
        for (int i = 0; i < 8; i++) {
            const int kk = hl * 16 + 2 * i;
            pf.u[i] = *reinterpret_cast<const unsigned int*>(&Ps[wid][l16][kk]);
        }

        // ---- O += P @ V (4 x 16-col output chunks, WMMA accumulate) ----
        #pragma unroll
        for (int j = 0; j < 4; j++) {
            AFrag vf;
            const int dv  = j * 16 + l16;
            const int kbb = hl * 16;
            #pragma unroll
            for (int i = 0; i < 8; i++)
                vf.u[i] = *reinterpret_cast<const unsigned int*>(&Vt[dv][kbb + 2 * i]);
            o[j] = __builtin_amdgcn_wmma_f32_16x16x32_bf16(
                       false, pf.v, false, vf.v, (short)0, o[j], false, false);
        }
        __syncthreads();
    }

    // ---- normalize + store, head-interleaved [B,T,H*64] ----
    #pragma unroll
    for (int j = 0; j < 4; j++) {
        const int dv = j * 16 + l16;
        #pragma unroll
        for (int r = 0; r < 8; r++) {
            const int row = q0 + hl * 8 + r;
            attn_out[((size_t)(b * 2048 + row)) * 1024 + hd * 64 + dv] =
                (bf16)(o[j][r] / l_r[r]);
        }
    }
}

// ---------------------------------------------------------------------------
// Host-side orchestration
// ---------------------------------------------------------------------------
extern "C" void kernel_launch(void* const* d_in, const int* in_sizes, int n_in,
                              void* d_out, int out_size, void* d_ws, size_t ws_size,
                              hipStream_t stream) {
    const float* x      = (const float*)d_in[0];
    const float* ln1_g  = (const float*)d_in[1];
    const float* ln1_b  = (const float*)d_in[2];
    const float* qkv_w  = (const float*)d_in[3];
    const float* qkv_b  = (const float*)d_in[4];
    const float* proj_w = (const float*)d_in[5];
    const float* proj_b = (const float*)d_in[6];
    const float* ln2_g  = (const float*)d_in[7];
    const float* ln2_b  = (const float*)d_in[8];
    const float* ff1_w  = (const float*)d_in[9];
    const float* ff1_b  = (const float*)d_in[10];
    const float* ff2_w  = (const float*)d_in[11];
    const float* ff2_b  = (const float*)d_in[12];
    float* out = (float*)d_out;

    const int M = 2 * 2048;   // B*T = 4096
    char* ws = (char*)d_ws;
    size_t off = 0;
    auto alloc = [&](size_t bytes) -> void* {
        void* p = ws + off;
        off += (bytes + 255) & ~(size_t)255;
        return p;
    };
    bf16*  wqkv  = (bf16*)alloc((size_t)1024 * 3072 * 2);
    bf16*  wproj = (bf16*)alloc((size_t)1024 * 1024 * 2);
    bf16*  wff1  = (bf16*)alloc((size_t)1024 * 4096 * 2);
    bf16*  wff2  = (bf16*)alloc((size_t)4096 * 1024 * 2);
    bf16*  h1    = (bf16*)alloc((size_t)M * 1024 * 2);
    bf16*  qkvb  = (bf16*)alloc((size_t)M * 3072 * 2);
    bf16*  attnb = (bf16*)alloc((size_t)M * 1024 * 2);
    float* x1    = (float*)alloc((size_t)M * 1024 * 4);
    bf16*  h2    = (bf16*)alloc((size_t)M * 1024 * 2);
    bf16*  ffh   = (bf16*)alloc((size_t)M * 4096 * 2);

    auto cvt = [&](const float* src, bf16* dst, int n) {
        cvt_f32_bf16_kernel<<<(n + 255) / 256, 256, 0, stream>>>(src, dst, n);
    };
    cvt(qkv_w,  wqkv,  1024 * 3072);
    cvt(proj_w, wproj, 1024 * 1024);
    cvt(ff1_w,  wff1,  1024 * 4096);
    cvt(ff2_w,  wff2,  4096 * 1024);

    // LN1 -> h1
    ln_bf16_kernel<<<M, 256, 0, stream>>>(x, ln1_g, ln1_b, h1, 1024);
    // QKV GEMM: [4096,1024] x [1024,3072] -> bf16
    gemm_bf16_kernel<0, true, false><<<dim3(24, 32), 256, 0, stream>>>(
        h1, wqkv, qkv_b, nullptr, qkvb, M, 3072, 1024);
    // causal flash attention
    attn_kernel<<<dim3(16, 32), 256, 0, stream>>>(qkvb, attnb);
    // proj GEMM + residual(x) -> x1 (fp32)
    gemm_bf16_kernel<0, false, true><<<dim3(8, 32), 256, 0, stream>>>(
        attnb, wproj, proj_b, x, x1, M, 1024, 1024);
    // LN2 -> h2
    ln_bf16_kernel<<<M, 256, 0, stream>>>(x1, ln2_g, ln2_b, h2, 1024);
    // FF1 GEMM + GELU -> bf16
    gemm_bf16_kernel<1, true, false><<<dim3(32, 32), 256, 0, stream>>>(
        h2, wff1, ff1_b, nullptr, ffh, M, 4096, 1024);
    // FF2 GEMM + residual(x1) -> out (fp32)
    gemm_bf16_kernel<0, false, true><<<dim3(8, 32), 256, 0, stream>>>(
        ffh, wff2, ff2_b, x1, out, M, 1024, 4096);
}
